// MDNDAR_72146860638892
// MI455X (gfx1250) — matrix operand order repaired and checked by reading, not compiled, not run on hardware
//
#include <hip/hip_runtime.h>
#include <stdint.h>

// ---------------------------------------------------------------------------
// MDN-DAR persistent-kernel implementation for gfx1250 (MI455X).
//   B=32, T=512, D_IN=256, H=1024, D_OUT=80, G=8
// bf16 WMMA (v_wmma_f32_16x16x32_bf16), weights pre-packed into wave32
// B-fragment layout, persistent kernel + 3 grid barriers per step,
// async global->LDS staging, double-buffered fragment pipeline.
// ---------------------------------------------------------------------------

typedef __attribute__((ext_vector_type(16))) __bf16        v16bf;
typedef __attribute__((ext_vector_type(8)))  float         v8f;
typedef __attribute__((ext_vector_type(4)))  unsigned int  v4u;
typedef __attribute__((__vector_size__(16))) int           v4i_t;

constexpr int kB    = 32;
constexpr int kT    = 512;
constexpr int kDIN  = 256;
constexpr int kH    = 1024;
constexpr int kDOUT = 80;
constexpr int kG    = 8;
constexpr int kGD   = kG * kDOUT;        // 640
constexpr int kK1   = kDIN + kDOUT;      // 336 (real concat width)
constexpr int kK1P  = 384;               // padded to even # of 32-wide k-tiles
constexpr int kKT1  = kK1P / 32;         // 12 k-tiles for the ih GEMM (even)
constexpr int kKTH  = kH / 32;           // 32 k-tiles for H-wide GEMMs (even)
constexpr int kNTH  = kH / 16;           // 64 n-tiles over H
constexpr int kNTGD = kGD / 16;          // 40 n-tiles over G*D_OUT

constexpr int kNWG        = 16;          // persistent workgroups
constexpr int kThreads    = 256;         // 8 waves (wave32)
constexpr int kWavesPerWG = kThreads / 32;
constexpr int kNWaves     = kNWG * kWavesPerWG;   // 128

constexpr int kLdaA      = (kK1P + kH) / 2;       // uints per LDS row, phase A (704)
constexpr int kLdaA4     = kLdaA / 4;             // v4u per LDS row (176)
constexpr int kSmemA     = kB * kLdaA * 4;        // 90112 B activation stage
constexpr int kSmemBytes = kSmemA + 2048;         // + logits scratch

// ---------------------------------------------------------------------------

__device__ __forceinline__ unsigned short f2bf(float f) {
  unsigned int u = __float_as_uint(f);
  unsigned int r = u + 0x7FFFu + ((u >> 16) & 1u);   // RNE
  return (unsigned short)(r >> 16);
}

__device__ __forceinline__ unsigned int pack2bf(float lo, float hi) {
  return (unsigned int)f2bf(lo) | ((unsigned int)f2bf(hi) << 16);
}

union BF16x16 {
  v16bf          v;
  unsigned short u[16];
  unsigned int   w[8];
  v4u            q[2];
};

// ---- gfx1250 async global->LDS staging (ASYNCcnt-tracked), with fallback ---
#if __has_builtin(__builtin_amdgcn_global_load_async_to_lds_b128)
#define HAS_ASYNC_LDS 1
#else
#define HAS_ASYNC_LDS 0
#endif

__device__ __forceinline__ void copy16_g2l(const v4u* g, v4u* l) {
#if HAS_ASYNC_LDS
  v4i_t* gp = (v4i_t*)(void*)const_cast<v4u*>(g);
  v4i_t* lp = (v4i_t*)(void*)l;
  __builtin_amdgcn_global_load_async_to_lds_b128(
      (__attribute__((address_space(1))) v4i_t*)gp,
      (__attribute__((address_space(3))) v4i_t*)lp,
      0, 0);
#else
  *l = *g;
#endif
}

__device__ __forceinline__ void wait_staging() {
#if HAS_ASYNC_LDS
#if __has_builtin(__builtin_amdgcn_s_wait_asynccnt)
  __builtin_amdgcn_s_wait_asynccnt(0);
#else
  asm volatile("s_wait_asynccnt 0" ::: "memory");
#endif
#endif
  __syncthreads();
}

// ---------------------------------------------------------------------------
// One 16x16 output tile; KT must be even. Double-buffered software pipeline:
// each WMMA consumes fragments loaded a full stage earlier (no register
// copies, no hazard-NOP moves), loop kept rolled to bound register pressure.
// ---------------------------------------------------------------------------
__device__ __forceinline__ void gemm_tile(const unsigned int* AU, int ldaU,
                                          int colOffU, int mbase,
                                          const unsigned short* __restrict__ pW,
                                          int KT, int lane, v8f& acc) {
  const int lHi = lane >> 4;
  // per-lane A base: row (mbase + lane&15), starting at this lane's 16B chunk
  const v4u* arow =
      (const v4u*)(AU + (mbase + (lane & 15)) * ldaU + colOffU + 8 * lHi);
  const v4u* bbase = reinterpret_cast<const v4u*>(pW) + lane * 2;

  BF16x16 a0, b0;
  b0.q[0] = bbase[0];
  b0.q[1] = bbase[1];
  a0.q[0] = arow[0];
  a0.q[1] = arow[1];
#pragma unroll 1
  for (int kt = 0; kt < KT; kt += 2) {
    BF16x16 a1, b1;
    b1.q[0] = bbase[(kt + 1) * 64 + 0];       // 512 halves = 64 v4u per k-tile
    b1.q[1] = bbase[(kt + 1) * 64 + 1];
    a1.q[0] = arow[(kt + 1) * 16];
    a1.q[1] = arow[(kt + 1) * 16 + 1];
    __builtin_prefetch((const void*)(pW + (size_t)(kt + 6) * 512 + lane * 16), 0, 1);
    acc = __builtin_amdgcn_wmma_f32_16x16x32_bf16(
        false, a0.v, false, b0.v, (short)0, acc, false, false);
    int kn = (kt + 2 < KT) ? (kt + 2) : 0;    // wrap: last prefetch unused
    b0.q[0] = bbase[kn * 64 + 0];
    b0.q[1] = bbase[kn * 64 + 1];
    a0.q[0] = arow[kn * 16];
    a0.q[1] = arow[kn * 16 + 1];
    acc = __builtin_amdgcn_wmma_f32_16x16x32_bf16(
        false, a1.v, false, b1.v, (short)0, acc, false, false);
  }
}

// Sense-reversing device-wide barrier over kNWG blocks.
__device__ __forceinline__ void grid_sync(int* bar) {
  __threadfence();
  __syncthreads();
  if (threadIdx.x == 0) {
    volatile int* vgen = (volatile int*)(bar + 1);
    int g = *vgen;
    if (atomicAdd(bar, 1) == kNWG - 1) {
      *(volatile int*)bar = 0;
      __threadfence();
      atomicAdd(bar + 1, 1);
    } else {
      while (*vgen == g) { __builtin_amdgcn_s_sleep(1); }
    }
  }
  __syncthreads();
  __threadfence();
}

// ---------------------------------------------------------------------------
// Weight packer: W (Nreal x Kreal, row-major f32) -> bf16 WMMA B-fragments.
// dst[(nt*KT + kt)*512 + lane*16 + j] = W[nt*16 + (lane&15), kt*32 + j + 16*(lane>>4)]
// ---------------------------------------------------------------------------
__global__ void mdndar_pack(const float* __restrict__ W, unsigned short* __restrict__ dst,
                            int Nreal, int Kreal, int NT, int KT) {
  int total  = NT * KT * 512;
  int stride = gridDim.x * blockDim.x;
  for (int i = blockIdx.x * blockDim.x + threadIdx.x; i < total; i += stride) {
    int nt   = i / (KT * 512);
    int rem  = i - nt * (KT * 512);
    int kt   = rem >> 9;
    int e    = rem & 511;
    int lane = e >> 4;
    int j    = e & 15;
    int n    = nt * 16 + (lane & 15);
    int k    = kt * 32 + j + 16 * (lane >> 4);
    float v  = (n < Nreal && k < Kreal) ? W[(size_t)n * Kreal + k] : 0.0f;
    dst[i] = f2bf(v);
  }
}

// h0 = 0, barrier counters = 0.
__global__ void mdndar_init(unsigned short* hbuf0, int* bar) {
  int i      = blockIdx.x * blockDim.x + threadIdx.x;
  int stride = gridDim.x * blockDim.x;
  if (i < 2) bar[i] = 0;
  for (int k = i; k < kB * kH; k += stride) hbuf0[k] = 0;
}

// ---------------------------------------------------------------------------
// Persistent main kernel: all T steps, 3 phases per step.
// ---------------------------------------------------------------------------
__global__ void __launch_bounds__(kThreads, 1)
mdndar_main(const float* __restrict__ x,
            const unsigned short* __restrict__ pWih,
            const unsigned short* __restrict__ pWhh,
            const unsigned short* __restrict__ pWlin,
            const unsigned short* __restrict__ pWsig,
            const unsigned short* __restrict__ pWmu,
            const unsigned short* __restrict__ pWpi,
            const float* __restrict__ b_ih, const float* __restrict__ b_hh,
            const float* __restrict__ b_lin, const float* __restrict__ b_pi,
            const float* __restrict__ b_sig, const float* __restrict__ b_mu,
            unsigned short* hbuf0, unsigned short* hbuf1,
            unsigned short* obuf,
            int* idxb, int* bar,
            float* out_logpi, float* out_logsig, float* out_mu) {
  extern __shared__ char smem[];
  unsigned int* AU = (unsigned int*)smem;
  v4u* lds4        = (v4u*)smem;
  float* ldsLogits = (float*)(smem + kSmemA);

  const int lane = threadIdx.x & 31;
  const int wave = threadIdx.x >> 5;
  const int gw   = blockIdx.x * kWavesPerWG + wave;   // 0..127
  const int lHi  = lane >> 4;

  for (int t = 0; t < kT; ++t) {
    const unsigned short* hR = (t & 1) ? hbuf1 : hbuf0;
    unsigned short*       hW = (t & 1) ? hbuf0 : hbuf1;

    // ---- stage A = [x_t | prev_mu | pad | h_prev] built directly in LDS ----
    {
      // x part: cols 0..255 -> uints 0..127 per row (pair-packed bf16)
      for (int i = threadIdx.x; i < kB * (kDIN / 2); i += kThreads) {
        int r = i >> 7, p = i & 127;
        const float* xp = x + ((size_t)r * kT + t) * kDIN + p * 2;
        AU[r * kLdaA + p] = pack2bf(xp[0], xp[1]);
      }
      // mu feedback: cols 256..335 -> uints 128..167 (zero at t==0)
      for (int i = threadIdx.x; i < kB * 40; i += kThreads) {
        int r = i / 40, p = i - r * 40;
        unsigned int pk = 0;
        if (t > 0) {
          const float* mp = out_mu + (size_t)(r * kT + (t - 1)) * kGD
                          + idxb[r] * kDOUT + p * 2;
          pk = pack2bf(mp[0], mp[1]);
        }
        AU[r * kLdaA + 128 + p] = pk;
      }
      // zero pad: cols 336..383 -> uints 168..191
      for (int i = threadIdx.x; i < kB * 24; i += kThreads) {
        int r = i / 24, p = i - r * 24;
        AU[r * kLdaA + 168 + p] = 0;
      }
      // h part: async global->LDS, 128 v4u per row at v4u offset 48
      const v4u* hu = (const v4u*)hR;
      for (int i = threadIdx.x; i < kB * 128; i += kThreads) {
        int r = i >> 7, c = i & 127;
        copy16_g2l(&hu[i], &lds4[r * kLdaA4 + 48 + c]);
      }
    }
    wait_staging();

    // ---- phase A: h_new = tanh(inp@Wih^T + h@Whh^T + b_ih + b_hh) ----
    {
      int nt = gw >> 1, mbase = (gw & 1) * 16;
      v8f acc = {0.f, 0.f, 0.f, 0.f, 0.f, 0.f, 0.f, 0.f};
      gemm_tile(AU, kLdaA, 0,        mbase, pWih + (size_t)nt * kKT1 * 512, kKT1, lane, acc);
      gemm_tile(AU, kLdaA, kK1P / 2, mbase, pWhh + (size_t)nt * kKTH * 512, kKTH, lane, acc);
      int n = nt * 16 + (lane & 15);
      float bv = b_ih[n] + b_hh[n];
#pragma unroll
      for (int r = 0; r < 8; ++r) {
        int m = mbase + r + 8 * lHi;
        hW[m * kH + n] = f2bf(tanhf(acc[r] + bv));
      }
    }
    grid_sync(bar);

    // ---- stage A = h_new ----
    {
      const v4u* hu = (const v4u*)hW;
      for (int i = threadIdx.x; i < kB * 128; i += kThreads)
        copy16_g2l(&hu[i], &lds4[i]);
    }
    wait_staging();

    // ---- phase B: o = h_new@Wlin^T + b_lin ----
    {
      int nt = gw >> 1, mbase = (gw & 1) * 16;
      v8f acc = {0.f, 0.f, 0.f, 0.f, 0.f, 0.f, 0.f, 0.f};
      gemm_tile(AU, kH / 2, 0, mbase, pWlin + (size_t)nt * kKTH * 512, kKTH, lane, acc);
      int n = nt * 16 + (lane & 15);
      float bv = b_lin[n];
#pragma unroll
      for (int r = 0; r < 8; ++r) {
        int m = mbase + r + 8 * lHi;
        obuf[m * kH + n] = f2bf(acc[r] + bv);
      }
    }
    grid_sync(bar);

    // ---- stage A = o ----
    {
      const v4u* ou = (const v4u*)obuf;
      for (int i = threadIdx.x; i < kB * 128; i += kThreads)
        copy16_g2l(&ou[i], &lds4[i]);
    }
    wait_staging();

    // ---- phase C: MDN heads (sigma: jobs 0..79, mu: 80..159, pi: 160) ----
    for (int j = gw; j < 161; j += kNWaves) {
      if (j < 80) {                       // log_sigma head
        int s = j >> 1, mbase = (j & 1) * 16;
        v8f acc = {0.f, 0.f, 0.f, 0.f, 0.f, 0.f, 0.f, 0.f};
        gemm_tile(AU, kH / 2, 0, mbase, pWsig + (size_t)s * kKTH * 512, kKTH, lane, acc);
        int n = s * 16 + (lane & 15);
        float bv = b_sig[n];
#pragma unroll
        for (int r = 0; r < 8; ++r) {
          int m = mbase + r + 8 * lHi;
          out_logsig[(size_t)(m * kT + t) * kGD + n] = acc[r] + bv;
        }
      } else if (j < 160) {               // mu head
        int jj = j - 80;
        int s = jj >> 1, mbase = (jj & 1) * 16;
        v8f acc = {0.f, 0.f, 0.f, 0.f, 0.f, 0.f, 0.f, 0.f};
        gemm_tile(AU, kH / 2, 0, mbase, pWmu + (size_t)s * kKTH * 512, kKTH, lane, acc);
        int n = s * 16 + (lane & 15);
        float bv = b_mu[n];
#pragma unroll
        for (int r = 0; r < 8; ++r) {
          int m = mbase + r + 8 * lHi;
          out_mu[(size_t)(m * kT + t) * kGD + n] = acc[r] + bv;
        }
      } else {                            // pi head + log_softmax + argmax
        v8f a0 = {0.f, 0.f, 0.f, 0.f, 0.f, 0.f, 0.f, 0.f};
        v8f a1 = {0.f, 0.f, 0.f, 0.f, 0.f, 0.f, 0.f, 0.f};
        gemm_tile(AU, kH / 2, 0, 0,  pWpi, kKTH, lane, a0);
        gemm_tile(AU, kH / 2, 0, 16, pWpi, kKTH, lane, a1);
        int g = lane & 15;
        if (g < kG) {
          float bb = b_pi[g];
#pragma unroll
          for (int r = 0; r < 8; ++r) {
            ldsLogits[(r + 8 * lHi) * kG + g]      = a0[r] + bb;
            ldsLogits[(16 + r + 8 * lHi) * kG + g] = a1[r] + bb;
          }
        }
        asm volatile("s_wait_dscnt 0" ::: "memory");   // same-wave LDS RAW
        int bb = lane;                                 // one batch row per lane
        float vals[kG];
        float mx = -3.4e38f;
#pragma unroll
        for (int g2 = 0; g2 < kG; ++g2) {
          vals[g2] = ldsLogits[bb * kG + g2];
          mx = fmaxf(mx, vals[g2]);
        }
        float ssum = 0.f;
#pragma unroll
        for (int g2 = 0; g2 < kG; ++g2) ssum += __expf(vals[g2] - mx);
        float lz = mx + __logf(ssum);
        int am = 0; float bmv = vals[0];
#pragma unroll
        for (int g2 = 1; g2 < kG; ++g2)
          if (vals[g2] > bmv) { bmv = vals[g2]; am = g2; }
#pragma unroll
        for (int g2 = 0; g2 < kG; ++g2)
          out_logpi[(size_t)(bb * kT + t) * kG + g2] = vals[g2] - lz;
        idxb[bb] = am;
      }
    }
    grid_sync(bar);   // publishes out_mu + idxb for next step's staging
  }
}

// ---------------------------------------------------------------------------
extern "C" void kernel_launch(void* const* d_in, const int* in_sizes, int n_in,
                              void* d_out, int out_size, void* d_ws, size_t ws_size,
                              hipStream_t stream) {
  (void)in_sizes; (void)n_in; (void)out_size; (void)ws_size;

  const float* x     = (const float*)d_in[0];
  const float* W_ih  = (const float*)d_in[1];
  const float* b_ih  = (const float*)d_in[2];
  const float* W_hh  = (const float*)d_in[3];
  const float* b_hh  = (const float*)d_in[4];
  const float* W_lin = (const float*)d_in[5];
  const float* b_lin = (const float*)d_in[6];
  const float* W_pi  = (const float*)d_in[7];
  const float* b_pi  = (const float*)d_in[8];
  const float* W_sig = (const float*)d_in[9];
  const float* b_sig = (const float*)d_in[10];
  const float* W_mu  = (const float*)d_in[11];
  const float* b_mu  = (const float*)d_in[12];

  char* ws = (char*)d_ws;
  size_t off = 0;
  auto carve = [&](size_t bytes) -> void* {
    void* p = (void*)(ws + off);
    off += (bytes + 255) & ~(size_t)255;
    return p;
  };
  int* bar              = (int*)carve(256);
  unsigned short* pWih  = (unsigned short*)carve((size_t)kNTH  * kKT1 * 512 * 2);
  unsigned short* pWhh  = (unsigned short*)carve((size_t)kNTH  * kKTH * 512 * 2);
  unsigned short* pWlin = (unsigned short*)carve((size_t)kNTH  * kKTH * 512 * 2);
  unsigned short* pWsig = (unsigned short*)carve((size_t)kNTGD * kKTH * 512 * 2);
  unsigned short* pWmu  = (unsigned short*)carve((size_t)kNTGD * kKTH * 512 * 2);
  unsigned short* pWpi  = (unsigned short*)carve((size_t)1     * kKTH * 512 * 2);
  unsigned short* hbuf0 = (unsigned short*)carve((size_t)kB * kH * 2);
  unsigned short* hbuf1 = (unsigned short*)carve((size_t)kB * kH * 2);
  unsigned short* obuf  = (unsigned short*)carve((size_t)kB * kH * 2);
  int* idxb             = (int*)carve(kB * sizeof(int));

  float* out_logpi  = (float*)d_out;
  float* out_logsig = out_logpi + (size_t)kB * kT * kG;
  float* out_mu     = out_logsig + (size_t)kB * kT * kG * kDOUT;

  // Pack all weights into WMMA B-fragment layout (bf16).
  mdndar_pack<<<dim3(256), dim3(256), 0, stream>>>(W_ih,  pWih,  kH,  kK1, kNTH,  kKT1);
  mdndar_pack<<<dim3(256), dim3(256), 0, stream>>>(W_hh,  pWhh,  kH,  kH,  kNTH,  kKTH);
  mdndar_pack<<<dim3(256), dim3(256), 0, stream>>>(W_lin, pWlin, kH,  kH,  kNTH,  kKTH);
  mdndar_pack<<<dim3(256), dim3(256), 0, stream>>>(W_sig, pWsig, kGD, kH,  kNTGD, kKTH);
  mdndar_pack<<<dim3(256), dim3(256), 0, stream>>>(W_mu,  pWmu,  kGD, kH,  kNTGD, kKTH);
  mdndar_pack<<<dim3(256), dim3(256), 0, stream>>>(W_pi,  pWpi,  kG,  kH,  1,     kKTH);
  mdndar_init<<<dim3(64),  dim3(256), 0, stream>>>(hbuf0, bar);

  (void)hipFuncSetAttribute(reinterpret_cast<const void*>(mdndar_main),
                            hipFuncAttributeMaxDynamicSharedMemorySize, kSmemBytes);
  mdndar_main<<<dim3(kNWG), dim3(kThreads), kSmemBytes, stream>>>(
      x, pWih, pWhh, pWlin, pWsig, pWmu, pWpi,
      b_ih, b_hh, b_lin, b_pi, b_sig, b_mu,
      hbuf0, hbuf1, obuf, idxb, bar,
      out_logpi, out_logsig, out_mu);
}
